// FactoredMultilevelDense_14929306321790
// MI455X (gfx1250) — compile-verified, matrix operand order checked by reading, not run on hardware
//
#include <hip/hip_runtime.h>
#include <math.h>
#include <stdint.h>

// Problem sizes (fixed by the reference)
#define BB 4096
#define INF 256
#define UU 128
#define LW 64
#define LB 32
#define KK (LW * INF)          // 16384 : GEMM K
#define KSPLIT 8
#define KCHUNK (KK / KSPLIT)   // 2048 (= 8 latent factors x 256 x-cols)
#define MTILE 128
#define NKT (KCHUNK / 32)      // 64 K-tiles per chunk
#define TILE_BYTES 8192        // one 32(K) x 128(N) bf16 tile

typedef __attribute__((ext_vector_type(16))) __bf16 bf16x16;
typedef __attribute__((ext_vector_type(8)))  float  v8f;

__device__ __forceinline__ float softplus_f(float v) {
    return (v > 20.0f) ? v : log1pf(expf(v));
}

// ---------------------------------------------------------------------------
// Kernel 1: zw_ws[b,l] = zw_mu[gid[b],l] + softplus(zw_sigma[gid[b],l])*eps_w[b,l]
// ---------------------------------------------------------------------------
__global__ void zw_sample_kernel(const float* __restrict__ zw_mu,
                                 const float* __restrict__ zw_sigma,
                                 const float* __restrict__ eps_w,
                                 const int*   __restrict__ gid,
                                 float* __restrict__ zw_ws) {
    int idx = blockIdx.x * blockDim.x + threadIdx.x;
    if (idx >= BB * LW) return;
    int b = idx >> 6;
    int l = idx & (LW - 1);
    int g = gid[b];
    float mu = zw_mu[g * LW + l];
    float sg = softplus_f(zw_sigma[g * LW + l]);
    zw_ws[idx] = fmaf(sg, eps_w[idx], mu);
}

// ---------------------------------------------------------------------------
// Kernel 2: repack Wf (f32 [16384,128]) into fragment-ordered bf16:
//   wfrag[kt][n][lane][j]  (kt:512, n:8, lane:32, j:16) -- 32B per lane is
//   exactly the v16bf B-operand fragment for WMMA tile (k0=kt*32, n0=n*16):
//     u  = n*16 + (lane&15)
//     k  = kt*32 + (lane>>4)*8 + (j<8 ? j : (j-8)+16)
// So GEMM staging is a contiguous 8KB copy and LDS reads are fragment loads.
// ---------------------------------------------------------------------------
__global__ __launch_bounds__(256) void wfrag_kernel(const float* __restrict__ Wf,
                                                    __bf16* __restrict__ wfrag) {
    __shared__ float tile[32][129];          // padded vs bank conflicts
    const int kt = blockIdx.x;               // 0..511
    const int t  = threadIdx.x;              // 0..255
    #pragma unroll
    for (int i = 0; i < 16; ++i) {           // coalesced 32x128 f32 load
        int idx = t + 256 * i;
        int r = idx >> 7, c = idx & 127;
        tile[r][c] = Wf[(size_t)(kt * 32 + r) * UU + c];
    }
    __syncthreads();
    const int lane  = t & 31;
    const int u     = ((t >> 5) * 16) + (lane & 15);
    const int khalf = (lane >> 4) * 8;
    bf16x16 v;
    #pragma unroll
    for (int j = 0; j < 16; ++j) {
        int kkl = khalf + ((j < 8) ? j : (j - 8) + 16);
        v[j] = (__bf16)tile[kkl][u];
    }
    *(bf16x16*)(wfrag + ((size_t)kt * 256 + t) * 16) = v;   // coalesced 32B/thread
}

// ---------------------------------------------------------------------------
// Kernel 3: out[b,u] = ((zb_mu + softplus(zb_sigma)*eps_b) @ Bf)[b,u]
// Also the deterministic initializer of d_out (harness poisons it).
// ---------------------------------------------------------------------------
__global__ void bias_kernel(const float* __restrict__ zb_mu,
                            const float* __restrict__ zb_sigma,
                            const float* __restrict__ eps_b,
                            const float* __restrict__ Bf,
                            const int*   __restrict__ gid,
                            float* __restrict__ out) {
    __shared__ float bs[LB];
    int b = blockIdx.x;
    int u = threadIdx.x;                     // 0..127
    if (u < LB) {
        int g = gid[b];
        bs[u] = fmaf(softplus_f(zb_sigma[g * LB + u]), eps_b[b * LB + u],
                     zb_mu[g * LB + u]);
    }
    __syncthreads();
    float acc = 0.0f;
    #pragma unroll
    for (int k = 0; k < LB; ++k)
        acc = fmaf(bs[k], Bf[k * UU + u], acc);
    out[(size_t)b * UU + u] = acc;
}

// ---------------------------------------------------------------------------
// Kernel 4: out += Y[4096,16384] @ WfK, with Y[b, l*256+i] = zw[b,l]*x[b,i].
// Grid (32 M-tiles, 8 K-chunks) x 256 threads (8 waves).
// B tile (8KB) staged once per block per K-step via async global->LDS
// (double buffered, ASYNCcnt + barrier), then each wave does 8 WMMAs.
// K-steps ordered (i outer, l inner) so one x segment is reused 8x with
// register-resident zw scalars. K-split partials land via f32 atomics.
// ---------------------------------------------------------------------------
__global__ __launch_bounds__(256) void gemm_kernel(
    const float*  __restrict__ x,
    const float*  __restrict__ zw_ws,
    const __bf16* __restrict__ wfrag,
    float*        __restrict__ out) {

    __shared__ __bf16 btile[2][TILE_BYTES / 2];   // 2 x 8KB double buffer

    const int tid   = threadIdx.x;
    const int lane  = tid & 31;
    const int wv    = tid >> 5;              // 0..7
    const int m0    = blockIdx.x * MTILE;
    const int kc    = blockIdx.y;            // K-chunk
    const int mlane = lane & 15;
    const int khalf = (lane >> 4) * 8;
    const int row   = m0 + wv * 16 + mlane;

    const uint32_t lbase = (uint32_t)(uintptr_t)&btile[0][0];
    const uint64_t gbase = (uint64_t)(uintptr_t)wfrag;

    // async-stage K-tile for logical iteration `it` into buffer (it&1).
    // iteration order: it = ii*8 + li  ->  k-tile kt = (kc*8 + li)*8 + ii
    auto stage = [&](int it) {
        int kt = (kc * 8 + (it & 7)) * 8 + (it >> 3);
        uint32_t la = lbase + (uint32_t)((it & 1) * TILE_BYTES + tid * 32);
        uint64_t ga = gbase + (uint64_t)kt * TILE_BYTES + (uint64_t)tid * 32;
        asm volatile(
            "global_load_async_to_lds_b128 %0, %1, off\n\t"
            "global_load_async_to_lds_b128 %0, %1, off offset:16"
            :: "v"(la), "v"(ga) : "memory");
    };

    v8f acc[8];
    const v8f vz = {0.f, 0.f, 0.f, 0.f, 0.f, 0.f, 0.f, 0.f};
    #pragma unroll
    for (int n = 0; n < 8; ++n) acc[n] = vz;

    // 8 latent-factor scales for this row's K-chunk, register resident
    float zw_l[8];
    #pragma unroll
    for (int li = 0; li < 8; ++li)
        zw_l[li] = zw_ws[(size_t)row * LW + kc * 8 + li];

    stage(0);

    #pragma unroll 1
    for (int ii = 0; ii < 8; ++ii) {
        // one 32-wide x segment, reused for all 8 latent factors
        const float* xp = x + (size_t)row * INF + ii * 32 + khalf;
        float4 xa0 = *(const float4*)(xp);
        float4 xa1 = *(const float4*)(xp + 4);
        float4 xb0 = *(const float4*)(xp + 16);
        float4 xb1 = *(const float4*)(xp + 20);

        #pragma unroll
        for (int li = 0; li < 8; ++li) {
            const int it = ii * 8 + li;
            asm volatile("s_wait_asynccnt 0x0" ::: "memory");
            __syncthreads();
            stage(it + 1);   // it+1==64 maps to a valid in-bounds dummy tile

            const float zws = zw_l[li];
            bf16x16 a;
            a[0]  = (__bf16)(zws * xa0.x); a[1]  = (__bf16)(zws * xa0.y);
            a[2]  = (__bf16)(zws * xa0.z); a[3]  = (__bf16)(zws * xa0.w);
            a[4]  = (__bf16)(zws * xa1.x); a[5]  = (__bf16)(zws * xa1.y);
            a[6]  = (__bf16)(zws * xa1.z); a[7]  = (__bf16)(zws * xa1.w);
            a[8]  = (__bf16)(zws * xb0.x); a[9]  = (__bf16)(zws * xb0.y);
            a[10] = (__bf16)(zws * xb0.z); a[11] = (__bf16)(zws * xb0.w);
            a[12] = (__bf16)(zws * xb1.x); a[13] = (__bf16)(zws * xb1.y);
            a[14] = (__bf16)(zws * xb1.z); a[15] = (__bf16)(zws * xb1.w);

            const __bf16* lb = &btile[it & 1][0];
            bf16x16 bfr[8];
            #pragma unroll
            for (int n = 0; n < 8; ++n)
                bfr[n] = *(const bf16x16*)(lb + (n * 32 + lane) * 16);
            #pragma unroll
            for (int n = 0; n < 8; ++n)
                acc[n] = __builtin_amdgcn_wmma_f32_16x16x32_bf16(
                    false, a, false, bfr[n], (short)0, acc[n], false, false);
        }
    }

    // C/D layout: VGPR r -> lanes 0-15: M=r, N=lane; lanes 16-31: M=8+r, N=lane-16
    const int rbase = m0 + wv * 16 + khalf;
    #pragma unroll
    for (int n = 0; n < 8; ++n) {
        #pragma unroll
        for (int r = 0; r < 8; ++r) {
            float* p = out + (size_t)(rbase + r) * UU + n * 16 + mlane;
            __hip_atomic_fetch_add(p, acc[n][r], __ATOMIC_RELAXED,
                                   __HIP_MEMORY_SCOPE_AGENT);
        }
    }
}

// ---------------------------------------------------------------------------
extern "C" void kernel_launch(void* const* d_in, const int* in_sizes, int n_in,
                              void* d_out, int out_size, void* d_ws, size_t ws_size,
                              hipStream_t stream) {
    (void)in_sizes; (void)n_in; (void)out_size; (void)ws_size;

    const float* x        = (const float*)d_in[0];   // [4096, 256]
    const float* eps_w    = (const float*)d_in[1];   // [4096, 64]
    const float* eps_b    = (const float*)d_in[2];   // [4096, 32]
    const float* zw_mu    = (const float*)d_in[3];   // [1000, 64]
    const float* zw_sigma = (const float*)d_in[4];   // [1000, 64]
    const float* Wf       = (const float*)d_in[5];   // [64, 32768] == [16384, 128]
    const float* zb_mu    = (const float*)d_in[6];   // [1000, 32]
    const float* zb_sigma = (const float*)d_in[7];   // [1000, 32]
    const float* Bf       = (const float*)d_in[8];   // [32, 128]
    const int*   gid      = (const int*)d_in[9];     // [4096]
    float*       out      = (float*)d_out;           // [4096, 128]

    // Workspace: zw_ws (1 MB f32) | wfrag (4 MB bf16, fragment-ordered)
    float*  zw_ws = (float*)d_ws;
    __bf16* wfrag = (__bf16*)((char*)d_ws + (size_t)BB * LW * sizeof(float));

    zw_sample_kernel<<<(BB * LW) / 256, 256, 0, stream>>>(zw_mu, zw_sigma,
                                                          eps_w, gid, zw_ws);
    wfrag_kernel<<<KK / 32, 256, 0, stream>>>(Wf, wfrag);
    bias_kernel<<<BB, UU, 0, stream>>>(zb_mu, zb_sigma, eps_b, Bf, gid, out);
    gemm_kernel<<<dim3(BB / MTILE, KSPLIT), 256, 0, stream>>>(x, zw_ws, wfrag, out);
}
